// MultiScaleVQVAE_32195074851359
// MI455X (gfx1250) — matrix operand order repaired and checked by reading, not compiled, not run on hardware
//
#include <hip/hip_runtime.h>
#include <hip/hip_bf16.h>
#include <math.h>

typedef __attribute__((ext_vector_type(16))) __bf16       v16bf;
typedef __attribute__((ext_vector_type(8)))  float        v8f;
typedef __attribute__((ext_vector_type(4)))  unsigned int v4u;
typedef __attribute__((ext_vector_type(2)))  unsigned int v2u;

// float -> bf16 (round-half-up)
static __device__ __forceinline__ __bf16 f2bf(float f) {
  unsigned int u = __builtin_bit_cast(unsigned int, f);
  unsigned short s = (unsigned short)((u + 0x8000u) >> 16);
  return __builtin_bit_cast(__bf16, s);
}
// pack two floats into one dword of bf16 (lo, hi)
static __device__ __forceinline__ unsigned int pack_bf2(float x, float y) {
  unsigned int ux = (__builtin_bit_cast(unsigned int, x) + 0x8000u) >> 16;
  unsigned int uy = (__builtin_bit_cast(unsigned int, y) + 0x8000u) & 0xFFFF0000u;
  return ux | uy;
}

struct BF16x16 { v4u lo, hi; };   // 32B, trivially copyable

// A fragment from LDS tile sA[rows][32]: per-lane rows = row, K runs
// [kbase,kbase+16) contiguous  ->  2x ds_load_b128
static __device__ __forceinline__ v16bf load_a_frag(const __bf16* sA, int row, int kbase) {
  BF16x16 t;
  t.lo = *(const v4u*)(sA + (row << 5) + kbase);
  t.hi = *(const v4u*)(sA + (row << 5) + kbase + 8);
  return __builtin_bit_cast(v16bf, t);
}
// B fragment from LDS: 16 fully contiguous bf16 -> 2x ds_load_b128
static __device__ __forceinline__ v16bf load_b_frag_lds(const __bf16* p) {
  BF16x16 t;
  t.lo = *(const v4u*)(p);
  t.hi = *(const v4u*)(p + 8);
  return __builtin_bit_cast(v16bf, t);
}
// B fragment from 16 contiguous global floats (4x b128), zeroed when !ok
static __device__ __forceinline__ void load16_bf(v16bf& b, const float* src, bool ok) {
  float vb[16];
#pragma unroll
  for (int t = 0; t < 4; ++t) {
    float4 q = ((const float4*)src)[t];
    vb[4 * t + 0] = q.x; vb[4 * t + 1] = q.y;
    vb[4 * t + 2] = q.z; vb[4 * t + 3] = q.w;
  }
#pragma unroll
  for (int e = 0; e < 16; ++e) b[e] = f2bf(ok ? vb[e] : 0.f);
}

// ---------------------------------------------------------------------------
// Implicit-GEMM conv, NHWC, register-blocked 2x4: each wave computes
// 32 Cout x 64 positions (8 accumulators); block's 8 waves share one 32x32
// bf16 weight tile in LDS. K ordered (tap, ci): each 16-K chunk is 16
// contiguous floats at one pixel -> 4x b128 + one branchless bounds check.
// All tile counts divide evenly; no guards; EXEC uniform at every WMMA.
// ---------------------------------------------------------------------------
template <int KSZ, bool RELU>
__global__ void __launch_bounds__(256)
conv_wmma_nhwc(const float* __restrict__ in, const float* __restrict__ wgt,
               const float* __restrict__ bias, float* __restrict__ out,
               int cinsh, int Cout, int ish, int osh, int stride, int pad,
               int Mtiles32) {
  __shared__ __bf16 sA[32 * 32];
  const int Cin  = 1 << cinsh;
  const int Kdim = (KSZ * KSZ) << cinsh;
  const int tid = threadIdx.x, lane = tid & 31, wave = tid >> 5;
  const int mt  = blockIdx.x % Mtiles32;
  const int ptg = blockIdx.x / Mtiles32;
  const int mbase = mt * 32;
  const int pw0 = (ptg * 8 + wave) * 64;
  const int Wi = 1 << ish, Wo = 1 << osh;
  const int ncol  = lane & 15;
  const int kbase = (lane >> 4) << 4;
  const int mhalf = (lane >> 4) << 3;

  int ohj[4], owj[4], nbsh[4];
#pragma unroll
  for (int j = 0; j < 4; ++j) {
    int pb = pw0 + j * 16 + ncol;
    owj[j] = pb & (Wo - 1);
    ohj[j] = (pb >> osh) & (Wo - 1);
    nbsh[j] = (pb >> (2 * osh)) << ish;
  }
  float bsv[2][8];
#pragma unroll
  for (int i = 0; i < 2; ++i)
#pragma unroll
    for (int rr = 0; rr < 8; ++rr) bsv[i][rr] = bias[mbase + i * 16 + mhalf + rr];

  v8f acc[2][4] = {};
  for (int k0 = 0; k0 < Kdim; k0 += 32) {
    __syncthreads();
    {  // stage 32x32 weight tile (OIHW -> (tap,ci) K-order), packed b64 store
      int e0 = tid << 2;
      int m = e0 >> 5, kl = e0 & 31;
      float w4[4];
#pragma unroll
      for (int q = 0; q < 4; ++q) {
        int kk = k0 + kl + q;
        int rs = kk >> cinsh, ci = kk & (Cin - 1);
        w4[q] = wgt[(mbase + m) * Kdim + ci * (KSZ * KSZ) + rs];
      }
      v2u pk = { pack_bf2(w4[0], w4[1]), pack_bf2(w4[2], w4[3]) };
      *(v2u*)(sA + e0) = pk;
      if (k0 + 32 < Kdim)
        __builtin_prefetch(&wgt[(mbase + m) * Kdim + kl + k0 + 32], 0, 1);
    }
    __syncthreads();

    v16bf a0 = load_a_frag(sA, ncol, kbase);
    v16bf a1 = load_a_frag(sA, ncol + 16, kbase);

    int kchunk = k0 + ((lane >> 4) << 4);
    int ci0 = kchunk & (Cin - 1);
    int rs = kchunk >> cinsh;
#pragma unroll
    for (int j = 0; j < 4; ++j) {
      int ih, iw;
      if (KSZ == 1) { ih = ohj[j]; iw = owj[j]; }        // only used s=1,p=0
      else {
        int r = rs / KSZ, s = rs - r * KSZ;
        ih = ohj[j] * stride - pad + r;
        iw = owj[j] * stride - pad + s;
      }
      bool ok = (KSZ == 1) ||
                (((unsigned)ih < (unsigned)Wi) && ((unsigned)iw < (unsigned)Wi));
      int base = ((((nbsh[j] + ih) << ish) + iw) << cinsh) + ci0;
      v16bf b;
      load16_bf(b, in + (ok ? base : 0), ok);
      acc[0][j] = __builtin_amdgcn_wmma_f32_16x16x32_bf16(false, a0, false, b,
                                                          (short)0, acc[0][j], false, false);
      acc[1][j] = __builtin_amdgcn_wmma_f32_16x16x32_bf16(false, a1, false, b,
                                                          (short)0, acc[1][j], false, false);
    }
  }

#pragma unroll
  for (int j = 0; j < 4; ++j) {
    int pb = pw0 + j * 16 + ncol;
#pragma unroll
    for (int i = 0; i < 2; ++i) {
      float vo[8];
#pragma unroll
      for (int rr = 0; rr < 8; ++rr) {
        float v = acc[i][j][rr] + bsv[i][rr];
        vo[rr] = RELU ? fmaxf(v, 0.f) : v;
      }
      float4* op = (float4*)(out + pb * Cout + mbase + i * 16 + mhalf);
      op[0] = make_float4(vo[0], vo[1], vo[2], vo[3]);
      op[1] = make_float4(vo[4], vo[5], vo[6], vo[7]);
    }
  }
}

// ---------------------------------------------------------------------------
// Transposed conv k=4,s=2,p=1, NHWC, parity classes (blockIdx.y), 2x4 blocked.
// Within class (ph,pw): taps kh=(1-ph)+2a, kw=(1-pw)+2b read in[u+ph-a,v+pw-b];
// K = 4*Cin ordered (ab, ci). Weights: (Cin, Cout, 4, 4).
// ---------------------------------------------------------------------------
__global__ void __launch_bounds__(256)
convt_wmma_nhwc(const float* __restrict__ in, const float* __restrict__ wgt,
                const float* __restrict__ bias, float* __restrict__ out,
                int cinsh, int Cout, int ish, int Mtiles32) {
  __shared__ __bf16 sA[32 * 32];
  const int Cin = 1 << cinsh;
  const int Kdim = Cin << 2;
  const int tid = threadIdx.x, lane = tid & 31, wave = tid >> 5;
  const int mt  = blockIdx.x % Mtiles32;
  const int ptg = blockIdx.x / Mtiles32;
  const int mbase = mt * 32;
  const int pw0 = (ptg * 8 + wave) * 64;
  const int ph = blockIdx.y & 1, pw = blockIdx.y >> 1;
  const int Wi = 1 << ish;
  const int osh = ish + 1;
  const int ncol  = lane & 15;
  const int kbase = (lane >> 4) << 4;
  const int mhalf = (lane >> 4) << 3;

  int uuj[4], vvj[4], nbsh[4];
#pragma unroll
  for (int j = 0; j < 4; ++j) {
    int pb = pw0 + j * 16 + ncol;
    vvj[j] = pb & (Wi - 1);
    uuj[j] = (pb >> ish) & (Wi - 1);
    nbsh[j] = (pb >> (2 * ish)) << ish;
  }
  float bsv[2][8];
#pragma unroll
  for (int i = 0; i < 2; ++i)
#pragma unroll
    for (int rr = 0; rr < 8; ++rr) bsv[i][rr] = bias[mbase + i * 16 + mhalf + rr];

  v8f acc[2][4] = {};
  for (int k0 = 0; k0 < Kdim; k0 += 32) {
    __syncthreads();
    {
      int e0 = tid << 2;
      int m = e0 >> 5, kl = e0 & 31;
      float w4[4];
#pragma unroll
      for (int q = 0; q < 4; ++q) {
        int kk = k0 + kl + q;
        int ab = kk >> cinsh, ci = kk & (Cin - 1);
        int kh = (1 - ph) + 2 * (ab >> 1), kw2 = (1 - pw) + 2 * (ab & 1);
        w4[q] = wgt[((ci * Cout + mbase + m) << 4) + (kh << 2) + kw2];
      }
      v2u pk = { pack_bf2(w4[0], w4[1]), pack_bf2(w4[2], w4[3]) };
      *(v2u*)(sA + e0) = pk;
    }
    __syncthreads();

    v16bf a0 = load_a_frag(sA, ncol, kbase);
    v16bf a1 = load_a_frag(sA, ncol + 16, kbase);

    int kchunk = k0 + ((lane >> 4) << 4);
    int ab = kchunk >> cinsh;
    int ci0 = kchunk & (Cin - 1);
#pragma unroll
    for (int j = 0; j < 4; ++j) {
      int ih = uuj[j] + ph - (ab >> 1);
      int iw = vvj[j] + pw - (ab & 1);
      bool ok = ((unsigned)ih < (unsigned)Wi) && ((unsigned)iw < (unsigned)Wi);
      int base = ((((nbsh[j] + ih) << ish) + iw) << cinsh) + ci0;
      v16bf b;
      load16_bf(b, in + (ok ? base : 0), ok);
      acc[0][j] = __builtin_amdgcn_wmma_f32_16x16x32_bf16(false, a0, false, b,
                                                          (short)0, acc[0][j], false, false);
      acc[1][j] = __builtin_amdgcn_wmma_f32_16x16x32_bf16(false, a1, false, b,
                                                          (short)0, acc[1][j], false, false);
    }
  }

#pragma unroll
  for (int j = 0; j < 4; ++j) {
    int oh = (uuj[j] << 1) + ph, ow = (vvj[j] << 1) + pw;
    int opix = (((nbsh[j] << 1) + oh) << osh) + ow;   // nb<<(ish+1) == nbsh<<1
#pragma unroll
    for (int i = 0; i < 2; ++i) {
      float vo[8];
#pragma unroll
      for (int rr = 0; rr < 8; ++rr) vo[rr] = fmaxf(acc[i][j][rr] + bsv[i][rr], 0.f);
      float4* op = (float4*)(out + opix * Cout + mbase + i * 16 + mhalf);
      op[0] = make_float4(vo[0], vo[1], vo[2], vo[3]);
      op[1] = make_float4(vo[4], vo[5], vo[6], vo[7]);
    }
  }
}

// ---------------------------------------------------------------------------
// VQ: argmin_c(||e_c||^2 - 2 x.e_c), x = z rows (flat [32768 x 256]).
// One wave owns 16 points (A fragments hoisted once); the block's 8 waves
// share each 16-code x 256-dim codebook tile staged as bf16 in LDS (8KB),
// so B fragments are 2x ds_load_b128. Shuffle-reduce min, tie -> lowest idx.
// ---------------------------------------------------------------------------
__global__ void __launch_bounds__(256)
vq_argmin_kernel(const float* __restrict__ z, const float* __restrict__ cb,
                 const float* __restrict__ cnorm, int* __restrict__ indices) {
  __shared__ __bf16 sB[16 * 256];
  const int tid = threadIdx.x, lane = tid & 31, wave = tid >> 5;
  const int pbase = (blockIdx.x * 8 + wave) * 16;
  const int pa = pbase + (lane & 15);
  const float* zrow = z + pa * 256;
  const int kb = (lane >> 4) << 4;

  v16bf afr[8];
#pragma unroll
  for (int kc = 0; kc < 8; ++kc) {
    float vb[16];
    int d0 = kc * 32 + kb;
#pragma unroll
    for (int t = 0; t < 2; ++t) {            // K runs [kb,kb+16) contiguous
      float4 q0 = *(const float4*)(zrow + d0 + 8 * t);
      float4 q1 = *(const float4*)(zrow + d0 + 8 * t + 4);
      vb[8 * t + 0] = q0.x; vb[8 * t + 1] = q0.y; vb[8 * t + 2] = q0.z; vb[8 * t + 3] = q0.w;
      vb[8 * t + 4] = q1.x; vb[8 * t + 5] = q1.y; vb[8 * t + 6] = q1.z; vb[8 * t + 7] = q1.w;
    }
#pragma unroll
    for (int e = 0; e < 16; ++e) afr[kc][e] = f2bf(vb[e]);
  }

  float bd[8]; int bi[8];
#pragma unroll
  for (int r = 0; r < 8; ++r) { bd[r] = 3.4e38f; bi[r] = 0; }

  for (int ct = 0; ct < 64; ++ct) {
    __syncthreads();
    {  // stage 16 codes x 256 dims as bf16: each thread packs 16 values
      int row = tid >> 4;
      int d0 = (tid & 15) << 4;
      const float* src = cb + (((ct << 4) + row) << 8) + d0;
      float4 q0 = ((const float4*)src)[0];
      float4 q1 = ((const float4*)src)[1];
      float4 q2 = ((const float4*)src)[2];
      float4 q3 = ((const float4*)src)[3];
      v4u p0 = { pack_bf2(q0.x, q0.y), pack_bf2(q0.z, q0.w),
                 pack_bf2(q1.x, q1.y), pack_bf2(q1.z, q1.w) };
      v4u p1 = { pack_bf2(q2.x, q2.y), pack_bf2(q2.z, q2.w),
                 pack_bf2(q3.x, q3.y), pack_bf2(q3.z, q3.w) };
      v4u* dst = (v4u*)(sB + (row << 8) + d0);
      dst[0] = p0; dst[1] = p1;
    }
    __syncthreads();

    v8f acc = {};
    const int code = (ct << 4) + (lane & 15);
    const __bf16* brow = sB + ((lane & 15) << 8);
#pragma unroll
    for (int kc = 0; kc < 8; ++kc) {
      v16bf b = load_b_frag_lds(brow + kc * 32 + ((lane >> 4) << 4));
      acc = __builtin_amdgcn_wmma_f32_16x16x32_bf16(false, afr[kc], false, b,
                                                    (short)0, acc, false, false);
    }
    float cn = cnorm[code];
#pragma unroll
    for (int r = 0; r < 8; ++r) {
      float dist = cn - 2.0f * acc[r];
      if (dist < bd[r]) { bd[r] = dist; bi[r] = code; }
    }
  }

#pragma unroll
  for (int r = 0; r < 8; ++r) {
#pragma unroll
    for (int off = 1; off < 16; off <<= 1) {   // stays within 16-lane halves
      float od = __shfl_xor(bd[r], off, 32);
      int   oi = __shfl_xor(bi[r], off, 32);
      if (od < bd[r] || (od == bd[r] && oi < bi[r])) { bd[r] = od; bi[r] = oi; }
    }
    if ((lane & 15) == 0)
      indices[pbase + r + ((lane >> 4) << 3)] = bi[r];
  }
}

// --------------------------- small direct kernels ---------------------------

// conv1: 1->64, k3 s1 p1; x [8,1,256,256]; out NHWC [8,256,256,64]
__global__ void conv1_relu_nhwc(const float* __restrict__ x, const float* __restrict__ w,
                                const float* __restrict__ b, float* __restrict__ out,
                                int total) {
  int o = blockIdx.x * blockDim.x + threadIdx.x;
  if (o >= total) return;
  int c = o & 63, ow = (o >> 6) & 255, oh = (o >> 14) & 255, n = o >> 22;
  float acc = b[c];
#pragma unroll
  for (int r = 0; r < 3; ++r)
#pragma unroll
    for (int s = 0; s < 3; ++s) {
      int ih = oh - 1 + r, iw = ow - 1 + s;
      if ((unsigned)ih < 256u && (unsigned)iw < 256u)
        acc += x[(n << 16) + (ih << 8) + iw] * w[c * 9 + r * 3 + s];
    }
  out[o] = fmaxf(acc, 0.f);
}

// final conv: 64->1, k3 s1 p1 + sigmoid; d3 NHWC [8,256,256,64]
__global__ void conv_out_sigmoid_kernel(const float* __restrict__ d3, const float* __restrict__ w,
                                        const float* __restrict__ b, float* __restrict__ out,
                                        int total) {
  int o = blockIdx.x * blockDim.x + threadIdx.x;
  if (o >= total) return;
  int ow = o & 255, oh = (o >> 8) & 255, n = o >> 16;
  float acc = b[0];
#pragma unroll
  for (int r = 0; r < 3; ++r)
#pragma unroll
    for (int s = 0; s < 3; ++s) {
      int ih = oh - 1 + r, iw = ow - 1 + s;
      if ((unsigned)ih < 256u && (unsigned)iw < 256u) {
        const float* px = d3 + (((((n << 8) + ih) << 8) + iw) << 6);
#pragma unroll
        for (int c4 = 0; c4 < 16; ++c4) {
          float4 q = ((const float4*)px)[c4];
          int ci = c4 << 2;
          acc += q.x * w[(ci + 0) * 9 + r * 3 + s] +
                 q.y * w[(ci + 1) * 9 + r * 3 + s] +
                 q.z * w[(ci + 2) * 9 + r * 3 + s] +
                 q.w * w[(ci + 3) * 9 + r * 3 + s];
        }
      }
    }
  out[o] = 1.f / (1.f + expf(-acc));
}

__global__ void codebook_norm_kernel(const float* __restrict__ cb, float* __restrict__ cn) {
  int c = blockIdx.x * blockDim.x + threadIdx.x;
  if (c >= 1024) return;
  float s = 0.f;
  for (int d4 = 0; d4 < 64; ++d4) {
    float4 q = ((const float4*)(cb + (c << 8)))[d4];
    s += q.x * q.x + q.y * q.y + q.z * q.z + q.w * q.w;
  }
  cn[c] = s;
}

// z_q gather: NCHW copy (d_out) + NHWC copy (decoder input)
__global__ void gather_zq_kernel(const int* __restrict__ idx, const float* __restrict__ cb,
                                 float* __restrict__ zq_nchw, float* __restrict__ zq_nhwc,
                                 int total4) {
  int o = blockIdx.x * blockDim.x + threadIdx.x;
  if (o >= total4) return;
  int p = o >> 6, d = (o & 63) << 2;
  int c = idx[p];
  float4 q = *(const float4*)(cb + (c << 8) + d);
  *(float4*)(zq_nhwc + (p << 8) + d) = q;
  int n = p >> 12, hw = p & 4095;
  zq_nchw[(((n << 8) + d + 0) << 12) + hw] = q.x;
  zq_nchw[(((n << 8) + d + 1) << 12) + hw] = q.y;
  zq_nchw[(((n << 8) + d + 2) << 12) + hw] = q.z;
  zq_nchw[(((n << 8) + d + 3) << 12) + hw] = q.w;
}

// ---------------------------------------------------------------------------

extern "C" void kernel_launch(void* const* d_in, const int* in_sizes, int n_in,
                              void* d_out, int out_size, void* d_ws, size_t ws_size,
                              hipStream_t stream) {
  const float* x   = (const float*)d_in[0];
  const float* ew1 = (const float*)d_in[1];
  const float* eb1 = (const float*)d_in[2];
  const float* ew2 = (const float*)d_in[3];
  const float* eb2 = (const float*)d_in[4];
  const float* ew3 = (const float*)d_in[5];
  const float* eb3 = (const float*)d_in[6];
  const float* ew4 = (const float*)d_in[7];
  const float* eb4 = (const float*)d_in[8];
  const float* cb  = (const float*)d_in[9];
  const float* dw1 = (const float*)d_in[10];
  const float* db1 = (const float*)d_in[11];
  const float* dw2 = (const float*)d_in[12];
  const float* db2 = (const float*)d_in[13];
  const float* dw3 = (const float*)d_in[14];
  const float* db3 = (const float*)d_in[15];
  const float* dw4 = (const float*)d_in[16];
  const float* db4 = (const float*)d_in[17];
  (void)in_sizes; (void)n_in; (void)out_size; (void)ws_size;

  // d_out = recon | z_q (NCHW) | indices (reference return order)
  float* outf    = (float*)d_out;
  float* recon   = outf;                                  // 8*1*256*256 = 524288
  float* zq_nchw = outf + 524288;                         // 8*256*64*64 = 8388608
  int*   indices = (int*)(outf + 524288 + 8388608);       // 8*64*64     = 32768

  float* ws = (float*)d_ws;
  size_t off = 0;
  float* h1  = ws + off; off += 33554432ull;  // NHWC [8,256,256,64]
  float* h2  = ws + off; off += 16777216ull;  // NHWC [8,128,128,128]
  float* h3  = ws + off; off += 8388608ull;   // NHWC [8,64,64,256]
  float* z   = ws + off; off += 8388608ull;   // flat [32768,256]
  float* zqh = ws + off; off += 8388608ull;   // NHWC z_q for decoder
  float* cn  = ws + off; off += 1024ull;      // codebook norms
  float* d1v = h3;                            // reuse (h3 dead after conv4)
  float* d2v = h2;                            // reuse
  float* d3v = h1;                            // reuse

  // ---- encoder ----
  conv1_relu_nhwc<<<131072, 256, 0, stream>>>(x, ew1, eb1, h1, 33554432);
  // 64->128 s2: Cout/32=4, P=131072 -> 256 groups of 512 positions
  conv_wmma_nhwc<3, true ><<<1024, 256, 0, stream>>>(h1, ew2, eb2, h2,
      /*cinsh*/6, 128, /*ish*/8, /*osh*/7, 2, 1, /*Mtiles32*/4);
  // 128->256 s2: Cout/32=8, P=32768 -> 64 groups
  conv_wmma_nhwc<3, true ><<<512, 256, 0, stream>>>(h2, ew3, eb3, h3,
      7, 256, 7, 6, 2, 1, 8);
  // 1x1 256->256 (no relu); output z = flat [p][256]
  conv_wmma_nhwc<1, false><<<512, 256, 0, stream>>>(h3, ew4, eb4, z,
      8, 256, 6, 6, 1, 0, 8);

  // ---- vector quantization ----
  codebook_norm_kernel<<<4, 256, 0, stream>>>(cb, cn);
  vq_argmin_kernel<<<256, 256, 0, stream>>>(z, cb, cn, indices);
  gather_zq_kernel<<<8192, 256, 0, stream>>>(indices, cb, zq_nchw, zqh, 2097152);

  // ---- decoder ----
  conv_wmma_nhwc<1, true><<<512, 256, 0, stream>>>(zqh, dw1, db1, d1v,
      8, 256, 6, 6, 1, 0, 8);
  // convT 256->128: Cout/32=4, per-class P=32768 -> 64 groups
  convt_wmma_nhwc<<<dim3(256, 4), 256, 0, stream>>>(d1v, dw2, db2, d2v,
      /*cinsh*/8, 128, /*ish*/6, /*Mtiles32*/4);
  // convT 128->64: Cout/32=2, per-class P=131072 -> 256 groups
  convt_wmma_nhwc<<<dim3(512, 4), 256, 0, stream>>>(d2v, dw3, db3, d3v,
      7, 64, 7, 2);
  conv_out_sigmoid_kernel<<<2048, 256, 0, stream>>>(d3v, dw4, db4, recon, 524288);
}
